// MemoryCell_64132451664294
// MI455X (gfx1250) — matrix operand order repaired and checked by reading, not compiled, not run on hardware
//
#include <hip/hip_runtime.h>
#include <hip/hip_bf16.h>
#include <math.h>

// ---------------------------------------------------------------------------
// MemoryCell on MI455X (gfx1250, wave32, WMMA).
// Chunked causal linear attention (chunk C=256):
//   Y_c = (Q_c@mem + tril(Q_cK_c^T)@V_c) / (Q_c.zeta + rowsum)
// All GEMMs use v_wmma_f32_16x16x32_f16 (f16 in, f32 accumulate); the
// 2048x2048 "mem" master stays f32 (L2-resident) with an f16 shadow.
// The RoPE-like cs(t,j) gate factor is precomputed into a [T][D/2] table so
// the GEMM epilogue stays light (one load + v_exp) and accumulators never
// spill to scratch.
// ---------------------------------------------------------------------------

typedef __attribute__((ext_vector_type(16))) _Float16 v16h;
typedef __attribute__((ext_vector_type(8)))  _Float16 v8h;
typedef __attribute__((ext_vector_type(8)))  float    v8f;

#define T_DIM 4096
#define D_DIM 2048
#define C_DIM 256   // chunk length
#define N_CHUNK (T_DIM / C_DIM)

// ---- WMMA fragment loaders (wave32 layouts per CDNA5 ISA 7.12.2) ----------
// A (16x32 f16): lanes 0-15 hold row M=lane with K 0..7 (v0-3) and 16..23
// (v4-7); lanes 16-31 hold the same row with K 8..15 and 24..31.
__device__ inline v16h load_fragA(const _Float16* __restrict__ base, int ld,
                                  int row0, int k0, int r, int hi) {
  const _Float16* p = base + (size_t)(row0 + r) * ld + k0 + hi * 8;
  union { v16h v; v8h h[2]; } u;
  u.h[0] = *(const v8h*)(p);        // K = k0+hi*8 .. +7
  u.h[1] = *(const v8h*)(p + 16);   // K = k0+16+hi*8 .. +7
  return u.v;
}
// B (32x16 f16): lane r<16 holds column N=n0+r with K k0..k0+7 and k0+16..k0+23;
// lane r+16 holds K k0+8..k0+15 and k0+24..k0+31 (same order as A).  Column of
// B == row of P (we compute A@P^T), so each lane reads two 8-half runs of P's row n0+r.
__device__ inline v16h load_fragB(const _Float16* __restrict__ base, int ld,
                                  int n0, int k0, int r, int hi) {
  const _Float16* p = base + (size_t)(n0 + r) * ld + k0 + hi * 8;
  union { v16h v; v8h h[2]; } u;
  u.h[0] = *(const v8h*)(p);
  u.h[1] = *(const v8h*)(p + 16);
  return u.v;
}

// ---------------------------------------------------------------------------
// Generic NT WMMA GEMM:  acc = A(row-major MxK f16) @ P(row-major NxK f16)^T.
// Block = 256 threads = 8 waves (2 in M x 4 in N); wave tile 64x32 (4x2 WMMA
// tiles); block tile 128x128.  M,N multiples of 128, K multiple of 32 (true
// for every call here).  Epilogue selected by `flag`:
//   0: Cf = acc                                    (f32 store)
//   1: Ch = sigmoid(acc * cs_tab[t][n/2] / 2048)   (gated f16: query / key;
//                                                   cs table passed via Ym/ldy)
//   2: Ch = acc                                    (f16 store: V)
//   3: tril mask (n<=m), store Cf (f32) + Ch (f16) (intra scores S)
//   4: Cf = (Ym + acc) / (svec[m] + rvec[m])       (final output rows)
//   5: Cacc += acc (f32 master), Ch = Cacc         (mem update + f16 shadow)
// ---------------------------------------------------------------------------
__global__ __launch_bounds__(256)
void gemm_nt_wmma(const _Float16* __restrict__ A, int lda,
                  const _Float16* __restrict__ B, int ldb,
                  float* __restrict__ Cf, _Float16* __restrict__ Ch,
                  float* __restrict__ Cacc, int ldc,
                  const float* __restrict__ Ym, int ldy,
                  const float* __restrict__ svec, const float* __restrict__ rvec,
                  int K, int flag, int row_base) {
  const int lane = threadIdx.x & 31;
  const int wave = threadIdx.x >> 5;
  const int wm = wave & 1;          // 2 waves along M
  const int wn = wave >> 1;         // 4 waves along N
  const int m_wave = blockIdx.y * 128 + wm * 64;
  const int n_wave = blockIdx.x * 128 + wn * 32;
  const int r  = lane & 15;
  const int hi = lane >> 4;

  v8f acc[4][2] = {};

  for (int k0 = 0; k0 < K; k0 += 32) {
    v16h af[4], bf[2];
#pragma unroll
    for (int i = 0; i < 4; ++i) af[i] = load_fragA(A, lda, m_wave + i * 16, k0, r, hi);
#pragma unroll
    for (int j = 0; j < 2; ++j) bf[j] = load_fragB(B, ldb, n_wave + j * 16, k0, r, hi);
#pragma unroll
    for (int i = 0; i < 4; ++i)
#pragma unroll
      for (int j = 0; j < 2; ++j)
        acc[i][j] = __builtin_amdgcn_wmma_f32_16x16x32_f16(
            false, af[i], false, bf[j], (short)0, acc[i][j], false, false);
  }

  const float DK = 1.0f / 2048.0f;
  // C/D layout: VGPR e, lanes 0-15 -> M=e, lanes 16-31 -> M=e+8; N = lane&15.
#pragma unroll
  for (int i = 0; i < 4; ++i)
#pragma unroll
    for (int j = 0; j < 2; ++j) {
      const int gm0 = m_wave + i * 16 + hi * 8;
      const int gn  = n_wave + j * 16 + r;
#pragma unroll
      for (int e = 0; e < 8; ++e) {
        const int gm = gm0 + e;
        float v = acc[i][j][e];
        const size_t idx = (size_t)gm * ldc + gn;
        if (flag == 0) {
          Cf[idx] = v;
        } else if (flag == 1) {
          // Ym = cs table [T][D/2], ldy = D/2
          float cs = Ym[(size_t)(row_base + gm) * ldy + (gn >> 1)];
          float z = v * cs * DK;
          Ch[idx] = (_Float16)(1.0f / (1.0f + __expf(-z)));
        } else if (flag == 2) {
          Ch[idx] = (_Float16)v;
        } else if (flag == 3) {
          if (gn > gm) v = 0.0f;           // causal mask (keep s <= t)
          Cf[idx] = v;
          Ch[idx] = (_Float16)v;
        } else if (flag == 4) {
          float den = svec[gm] + rvec[gm];
          Cf[idx] = (Ym[(size_t)gm * ldy + gn] + v) / den;
        } else {
          float nv = Cacc[idx] + v;
          Cacc[idx] = nv;
          Ch[idx] = (_Float16)nv;
        }
      }
    }
}

// ---------------------------------------------------------------------------
// Small helper kernels
// ---------------------------------------------------------------------------
// cs_tab[t][i] = cos(t*theta_i) + sin(t*theta_i), theta_i = 10000^(-2i).
// One-time; keeps accurate libm trig (large-argument reduction) out of the
// GEMM epilogue.
__global__ void cs_table_kernel(float* __restrict__ tab, int T, int halfD) {
  size_t n = (size_t)T * halfD;
  size_t idx = (size_t)blockIdx.x * blockDim.x + threadIdx.x;
  size_t stride = (size_t)gridDim.x * blockDim.x;
  for (; idx < n; idx += stride) {
    int t = (int)(idx / halfD);
    int i = (int)(idx % halfD);
    float theta = powf(1.0e4f, -2.0f * (float)i);
    float ang = (float)t * theta;
    tab[idx] = cosf(ang) + sinf(ang);
  }
}

__global__ void f32_to_f16_kernel(const float* __restrict__ in,
                                  _Float16* __restrict__ out, size_t n) {
  size_t i = (size_t)blockIdx.x * blockDim.x + threadIdx.x;
  size_t stride = (size_t)gridDim.x * blockDim.x;
  for (; i < n; i += stride) out[i] = (_Float16)in[i];
}

__global__ void zero_f32_kernel(float* __restrict__ p, size_t n) {
  size_t i = (size_t)blockIdx.x * blockDim.x + threadIdx.x;
  size_t stride = (size_t)gridDim.x * blockDim.x;
  for (; i < n; i += stride) p[i] = 0.0f;
}

__global__ void zero_f16_kernel(_Float16* __restrict__ p, size_t n) {
  size_t i = (size_t)blockIdx.x * blockDim.x + threadIdx.x;
  size_t stride = (size_t)gridDim.x * blockDim.x;
  for (; i < n; i += stride) p[i] = (_Float16)0.0f;
}

// Transpose f16 [R][C] -> [C][R] via LDS tile (R,C multiples of 32).
__global__ __launch_bounds__(256)
void transpose_f16_kernel(const _Float16* __restrict__ in,
                          _Float16* __restrict__ out, int R, int C) {
  __shared__ _Float16 tile[32][33];
  const int bx = blockIdx.x * 32;   // column block (input)
  const int by = blockIdx.y * 32;   // row block (input)
  const int tx = threadIdx.x & 31;
  const int ty = threadIdx.x >> 5;  // 8 rows per pass
  for (int i = ty; i < 32; i += 8)
    tile[i][tx] = in[(size_t)(by + i) * C + (bx + tx)];
  __syncthreads();
  for (int i = ty; i < 32; i += 8)
    out[(size_t)(bx + i) * R + (by + tx)] = tile[tx][i];
}

// scale[t] = dot(Q[t,:], zeta)   (one block per row)
__global__ __launch_bounds__(256)
void dot_zeta_kernel(const _Float16* __restrict__ Q, const float* __restrict__ zeta,
                     float* __restrict__ out, int d) {
  __shared__ float red[256];
  const int t = blockIdx.x;
  float s = 0.0f;
  for (int i = threadIdx.x; i < d; i += 256)
    s += (float)Q[(size_t)t * d + i] * zeta[i];
  red[threadIdx.x] = s;
  __syncthreads();
  for (int off = 128; off > 0; off >>= 1) {
    if (threadIdx.x < off) red[threadIdx.x] += red[threadIdx.x + off];
    __syncthreads();
  }
  if (threadIdx.x == 0) out[t] = red[0];
}

// rowsum[t] = sum_s S[t][s]  (S already causal-masked)
__global__ void rowsum_kernel(const float* __restrict__ S, float* __restrict__ out, int C) {
  int t = blockIdx.x * blockDim.x + threadIdx.x;
  if (t < C) {
    float s = 0.0f;
    for (int j = 0; j < C; ++j) s += S[(size_t)t * C + j];
    out[t] = s;
  }
}

// zeta[i] += sum_t Kc[t][i]
__global__ void zeta_update_kernel(const _Float16* __restrict__ Kc,
                                   float* __restrict__ zeta, int C, int d) {
  int i = blockIdx.x * blockDim.x + threadIdx.x;
  if (i < d) {
    float s = zeta[i];
    for (int t = 0; t < C; ++t) s += (float)Kc[(size_t)t * d + i];
    zeta[i] = s;
  }
}

// ---------------------------------------------------------------------------
extern "C" void kernel_launch(void* const* d_in, const int* in_sizes, int n_in,
                              void* d_out, int out_size, void* d_ws, size_t ws_size,
                              hipStream_t stream) {
  (void)in_sizes; (void)n_in; (void)out_size; (void)ws_size;
  const float* x  = (const float*)d_in[0];
  const float* Wq = (const float*)d_in[1];
  const float* Wk = (const float*)d_in[2];
  const float* Wv = (const float*)d_in[3];
  float* out = (float*)d_out;

  const int T = T_DIM, D = D_DIM, C = C_DIM;

  char* w = (char*)d_ws;
  auto carve = [&](size_t bytes) {
    char* p = w;
    w += (bytes + 255) & ~(size_t)255;
    return (void*)p;
  };
  _Float16* xh    = (_Float16*)carve((size_t)T * D * 2);
  _Float16* Wqh   = (_Float16*)carve((size_t)D * D * 2);
  _Float16* Wkh   = (_Float16*)carve((size_t)D * D * 2);
  _Float16* Wvh   = (_Float16*)carve((size_t)D * D * 2);
  _Float16* query = (_Float16*)carve((size_t)T * D * 2);  // sigmoid-gated Q
  _Float16* keyg  = (_Float16*)carve((size_t)T * D * 2);  // sigmoid-gated K
  _Float16* Vh    = (_Float16*)carve((size_t)T * D * 2);
  float*    M2f   = (float*)   carve((size_t)D * D * 4);  // mem^T master (f32)
  _Float16* M2h   = (_Float16*)carve((size_t)D * D * 2);  // mem^T f16 shadow
  float*    cstab = (float*)   carve((size_t)T * (D / 2) * 4); // cs(t, j/2)
  float*    zeta  = (float*)   carve((size_t)D * 4);
  float*    Yint  = (float*)   carve((size_t)C * D * 4);
  float*    sinter= (float*)   carve((size_t)C * 4);
  float*    Sf    = (float*)   carve((size_t)C * C * 4);
  _Float16* Sh    = (_Float16*)carve((size_t)C * C * 2);
  float*    rsum  = (float*)   carve((size_t)C * 4);
  _Float16* Vt    = (_Float16*)carve((size_t)D * C * 2);
  _Float16* Kt    = (_Float16*)carve((size_t)D * C * 2);

  // --- one-time precompute / downconvert ---
  cs_table_kernel<<<2048, 256, 0, stream>>>(cstab, T, D / 2);
  f32_to_f16_kernel<<<2048, 256, 0, stream>>>(x,  xh,  (size_t)T * D);
  f32_to_f16_kernel<<<2048, 256, 0, stream>>>(Wq, Wqh, (size_t)D * D);
  f32_to_f16_kernel<<<2048, 256, 0, stream>>>(Wk, Wkh, (size_t)D * D);
  f32_to_f16_kernel<<<2048, 256, 0, stream>>>(Wv, Wvh, (size_t)D * D);
  zero_f32_kernel<<<2048, 256, 0, stream>>>(M2f, (size_t)D * D);
  zero_f16_kernel<<<2048, 256, 0, stream>>>(M2h, (size_t)D * D);
  zero_f32_kernel<<<1, 256, 0, stream>>>(zeta, (size_t)D);

  // --- projections with fused gate epilogue:  query/key = sigmoid(xW^T*cs/d)
  dim3 gProj(D / 128, T / 128);
  gemm_nt_wmma<<<gProj, 256, 0, stream>>>(xh, D, Wqh, D, nullptr, query, nullptr, D,
                                          cstab, D / 2, nullptr, nullptr, D, 1, 0);
  gemm_nt_wmma<<<gProj, 256, 0, stream>>>(xh, D, Wkh, D, nullptr, keyg, nullptr, D,
                                          cstab, D / 2, nullptr, nullptr, D, 1, 0);
  gemm_nt_wmma<<<gProj, 256, 0, stream>>>(xh, D, Wvh, D, nullptr, Vh, nullptr, D,
                                          nullptr, 0, nullptr, nullptr, D, 2, 0);

  // --- chunked causal linear attention; stream order serializes chunks ---
  for (int c = 0; c < N_CHUNK; ++c) {
    const size_t base = (size_t)c * C;
    const _Float16* Qc = query + base * D;
    const _Float16* Kc = keyg  + base * D;
    const _Float16* Vc = Vh    + base * D;

    // inter: Yint = Qc @ mem  (mem^T f16 shadow, rows indexed by output col)
    gemm_nt_wmma<<<dim3(D / 128, C / 128), 256, 0, stream>>>(
        Qc, D, M2h, D, Yint, nullptr, nullptr, D,
        nullptr, 0, nullptr, nullptr, D, 0, 0);
    // sinter[t] = Qc[t] . zeta_prev
    dot_zeta_kernel<<<C, 256, 0, stream>>>(Qc, zeta, sinter, D);
    // intra scores: S = tril(Qc @ Kc^T)  -> f32 + f16
    gemm_nt_wmma<<<dim3(C / 128, C / 128), 256, 0, stream>>>(
        Qc, D, Kc, D, Sf, Sh, nullptr, C,
        nullptr, 0, nullptr, nullptr, D, 3, 0);
    rowsum_kernel<<<1, 256, 0, stream>>>(Sf, rsum, C);
    // transposed chunk copies for S@V and the mem update
    transpose_f16_kernel<<<dim3(D / 32, C / 32), 256, 0, stream>>>(Vc, Vt, C, D);
    transpose_f16_kernel<<<dim3(D / 32, C / 32), 256, 0, stream>>>(Kc, Kt, C, D);
    // final rows: out = (Yint + S@Vc) / (sinter + rsum)
    gemm_nt_wmma<<<dim3(D / 128, C / 128), 256, 0, stream>>>(
        Sh, C, Vt, C, out + base * D, nullptr, nullptr, D,
        Yint, D, sinter, rsum, C, 4, 0);
    // state update: mem^T += Vc^T @ Kc  (f32 master + f16 shadow)
    gemm_nt_wmma<<<dim3(D / 128, D / 128), 256, 0, stream>>>(
        Vt, C, Kt, C, nullptr, M2h, M2f, D,
        nullptr, 0, nullptr, nullptr, C, 5, 0);
    zeta_update_kernel<<<D / 256, 256, 0, stream>>>(Kc, zeta, C, D);
  }
}
